// MERABlock_28071906246840
// MI455X (gfx1250) — matrix operand order repaired and checked by reading, not compiled, not run on hardware
//
#include <hip/hip_runtime.h>

// ---------------------------------------------------------------------------
// MERA block for MI455X (gfx1250, wave32). All matmul-like ops are lowered to
// a bf16 WMMA GEMM (v_wmma_f32_16x16x32_bf16, f32 accumulate) with LDS-staged,
// double-buffered tiles fed by CDNA5 async global->LDS copies (ASYNCcnt).
// The serial MPS recurrence runs as 32 blocks with the 256x256 A matrix staged
// in LDS (row stride 257 to avoid 64-bank conflicts).
// ---------------------------------------------------------------------------

typedef __bf16 bf16_t;
typedef __attribute__((ext_vector_type(16))) __bf16 v16bf;
typedef __attribute__((ext_vector_type(8)))  __bf16 v8bf;
typedef __attribute__((ext_vector_type(8)))  float  v8f;
typedef int v4i __attribute__((vector_size(16)));

#define Dm   1024
#define Gg   8
#define Hh   8
#define Kk   256
#define DHh  128
#define DFFc 4096
#define Bn   4
#define T0   4096
#define EPSf 1e-5f

// GEMM tiling
#define BM   128
#define BN   128
#define BK   64
#define PADK 72          // padded LDS row length in bf16 (144 B, 16B-aligned)

__device__ __forceinline__ bf16_t f2bf(float f) {
  unsigned u = __float_as_uint(f);
  unsigned r = (u + 0x7FFFu + ((u >> 16) & 1u)) >> 16;
  unsigned short s = (unsigned short)r;
  bf16_t b;
  __builtin_memcpy(&b, &s, 2);
  return b;
}

// 16-byte global -> LDS copy. Prefer the CDNA5 async-to-LDS path (ASYNCcnt),
// fall back to load+ds_store if the builtin is unavailable.
// Builtin signature (from hipcc diagnostic): arg0 = AS(1) v4i*, arg1 = AS(3)
// v4i*, then two int immediates (offset, cpol).
__device__ __forceinline__ void async16(const bf16_t* g, bf16_t* l) {
#if defined(__AMDGCN__) && __has_builtin(__builtin_amdgcn_global_load_async_to_lds_b128)
  __builtin_amdgcn_global_load_async_to_lds_b128(
      (__attribute__((address_space(1))) v4i*)g,
      (__attribute__((address_space(3))) v4i*)l, 0, 0);
#else
  *(v8bf*)l = *(const v8bf*)g;
#endif
}

template <int N>
__device__ __forceinline__ void wait_async() {
#if defined(__AMDGCN__) && __has_builtin(__builtin_amdgcn_s_wait_asynccnt)
  __builtin_amdgcn_s_wait_asynccnt(N);
#elif defined(__AMDGCN__)
  asm volatile("s_wait_asynccnt %0" ::"i"(N) : "memory");
#endif
}

// ---------------------------------------------------------------------------
// GEMM: C[m][n] = sum_k A[m][k] * Wt[n][k] (+ bias[n]), batched over z.
//   A  : bf16, row-major, leading dim lda, + z*aZ
//   Wt : bf16, OUTPUT-major (N rows of K contiguous), + z*bZ
//   Cf : f32 out (optional), Cbf: bf16 out (optional), both ldc, + z*cZ
// EPI: 0 = none, 1 = gelu(tanh approx)
// Block: 256 threads = 8 waves (4 M x 2 N). Block tile 128x128, K-tile 64,
// double-buffered LDS staged via async copies. Wave tile 32x64 = 8 WMMA accs.
// Requires M%128==0, N%128==0, Kd%64==0 (guaranteed by construction).
// ---------------------------------------------------------------------------
template <int EPI>
__global__ __launch_bounds__(256)
void gemm_wmma(const bf16_t* __restrict__ A, const bf16_t* __restrict__ Wt,
               const float* __restrict__ bias, float* __restrict__ Cf,
               bf16_t* __restrict__ Cbf,
               int M, int N, int Kd, int lda, int ldc,
               long aZ, long bZ, long cZ, long biasZ)
{
  extern __shared__ __align__(16) char smem[];
  bf16_t* sA = (bf16_t*)smem;                       // [2][BM*PADK]
  bf16_t* sB = sA + 2 * BM * PADK;                  // [2][BN*PADK]

  const int z = blockIdx.z;
  A  += (long)z * aZ;
  Wt += (long)z * bZ;
  const long cOff = (long)z * cZ;

  const int tid   = threadIdx.x;
  const int lane  = tid & 31;
  const int wave  = tid >> 5;
  const int waveM = wave & 3;    // 0..3 -> 32-row sub-tile
  const int waveN = wave >> 2;   // 0..1 -> 64-col sub-tile
  const int m0 = blockIdx.y * BM;
  const int n0 = blockIdx.x * BN;

  // Stage one 128x64 A tile + 128x64 B tile into LDS buffer `buf`.
  // 1024 16B chunks per matrix / 256 threads = 4 async issues each.
  auto issue_tile = [&](int buf, int k0) {
    bf16_t* dA = sA + buf * BM * PADK;
    bf16_t* dB = sB + buf * BN * PADK;
#pragma unroll
    for (int i = 0; i < 4; ++i) {
      int c = tid + i * 256;
      int row = c >> 3, ch = c & 7;
      async16(A + (long)(m0 + row) * lda + k0 + ch * 8, dA + row * PADK + ch * 8);
    }
#pragma unroll
    for (int i = 0; i < 4; ++i) {
      int c = tid + i * 256;
      int row = c >> 3, ch = c & 7;
      async16(Wt + (long)(n0 + row) * Kd + k0 + ch * 8, dB + row * PADK + ch * 8);
    }
  };

  const int mA  = lane & 15;
  const int hiA = (lane >> 4) * 8;
  const int nB  = lane & 15;
  const int hiB = (lane >> 4) * 8;

  v8f zero = {0.f, 0.f, 0.f, 0.f, 0.f, 0.f, 0.f, 0.f};
  v8f acc[8];
#pragma unroll
  for (int i = 0; i < 8; ++i) acc[i] = zero;

  const int nt = Kd / BK;
  issue_tile(0, 0);

  for (int t = 0; t < nt; ++t) {
    if (t + 1 < nt) {
      issue_tile((t + 1) & 1, (t + 1) * BK);
      wait_async<8>();     // current tile's 8 issues done; next 8 may fly
    } else {
      wait_async<0>();
    }
    __syncthreads();

    const bf16_t* ta = sA + (t & 1) * BM * PADK;
    const bf16_t* tb = sB + (t & 1) * BN * PADK;

#pragma unroll
    for (int kk = 0; kk < BK; kk += 32) {
      v16bf af[2];
#pragma unroll
      for (int p = 0; p < 2; ++p) {
        const bf16_t* ap = ta + (waveM * 32 + p * 16 + mA) * PADK + kk + hiA;
        v8bf lo = *(const v8bf*)ap;
        v8bf hi = *(const v8bf*)(ap + 16);
#pragma unroll
        for (int i = 0; i < 8; ++i) { af[p][i] = lo[i]; af[p][i + 8] = hi[i]; }
      }
#pragma unroll
      for (int j = 0; j < 4; ++j) {
        const bf16_t* bp = tb + (waveN * 64 + j * 16 + nB) * PADK + kk + hiB;
        v8bf l0 = *(const v8bf*)bp;
        v8bf l1 = *(const v8bf*)(bp + 16);
        v16bf bfrag;
#pragma unroll
        for (int i = 0; i < 8; ++i) { bfrag[i] = l0[i]; bfrag[i + 8] = l1[i]; }
        acc[j] = __builtin_amdgcn_wmma_f32_16x16x32_bf16(
            false, af[0], false, bfrag, (short)0, acc[j], false, false);
        acc[4 + j] = __builtin_amdgcn_wmma_f32_16x16x32_bf16(
            false, af[1], false, bfrag, (short)0, acc[4 + j], false, false);
      }
    }
    __syncthreads();   // protect buffer reuse at next iteration's issue
  }

  // Epilogue. C/D layout: VGPR r -> M = base + (lane>>4)*8 + r, N = lane&15.
  const int nOut = lane & 15;
#pragma unroll
  for (int p = 0; p < 2; ++p) {
    const int mBase = m0 + waveM * 32 + p * 16 + (lane >> 4) * 8;
#pragma unroll
    for (int j = 0; j < 4; ++j) {
      const v8f& a = acc[p * 4 + j];
      const int nCol = n0 + waveN * 64 + j * 16 + nOut;
      float bv = bias ? bias[biasZ * z + nCol] : 0.f;
#pragma unroll
      for (int r = 0; r < 8; ++r) {
        float v = a[r] + bv;
        if (EPI == 1) { // gelu tanh approx (jax.nn.gelu default)
          float c = v + 0.044715f * v * v * v;
          v = 0.5f * v * (1.f + tanhf(0.7978845608f * c));
        }
        long idx = cOff + (long)(mBase + r) * ldc + nCol;
        if (Cf)  Cf[idx]  = v;
        if (Cbf) Cbf[idx] = f2bf(v);
      }
    }
  }
}

// ---------------------------------------------------------------------------
// LayerNorm over D=1024 per row, optional residual add (out = LN(in+res)).
// in == out aliasing is safe (reads complete before writes per row/block).
// ---------------------------------------------------------------------------
__global__ __launch_bounds__(256)
void ln_kernel(float* __restrict__ out, const float* __restrict__ in,
               const float* __restrict__ res,
               const float* __restrict__ g, const float* __restrict__ b)
{
  const long row = blockIdx.x;
  const float* x = in + row * Dm;
  const float* r = res ? res + row * Dm : nullptr;
  const int t = threadIdx.x;
  float v[4], s = 0.f, s2 = 0.f;
#pragma unroll
  for (int i = 0; i < 4; ++i) {
    int c = t + i * 256;
    float xv = x[c];
    if (r) xv += r[c];
    v[i] = xv; s += xv; s2 += xv * xv;
  }
  __shared__ float sh[256], sh2[256];
  sh[t] = s; sh2[t] = s2; __syncthreads();
  for (int o = 128; o > 0; o >>= 1) {
    if (t < o) { sh[t] += sh[t + o]; sh2[t] += sh2[t + o]; }
    __syncthreads();
  }
  float mean = sh[0] * (1.f / Dm);
  float var  = sh2[0] * (1.f / Dm) - mean * mean;
  float inv  = rsqrtf(var + EPSf);
#pragma unroll
  for (int i = 0; i < 4; ++i) {
    int c = t + i * 256;
    out[row * Dm + c] = (v[i] - mean) * inv * g[c] + b[c];
  }
}

// ---------------------------------------------------------------------------
// Elementwise kernels
// ---------------------------------------------------------------------------
__global__ void sigmoid_mul_kernel(float* __restrict__ out,
                                   const float* __restrict__ a,
                                   const float* __restrict__ bsrc, long n)
{
  long i = (long)blockIdx.x * blockDim.x + threadIdx.x;
  if (i >= n) return;
  float g = 1.f / (1.f + expf(-a[i]));
  out[i] = g * bsrc[i];
}

__global__ void add2_kernel(float* __restrict__ out, const float* __restrict__ a,
                            const float* __restrict__ bsrc, long n)
{
  long i = (long)blockIdx.x * blockDim.x + threadIdx.x;
  if (i >= n) return;
  out[i] = a[i] + bsrc[i];
}

__global__ void cvt_bf_kernel(bf16_t* __restrict__ out,
                              const float* __restrict__ in, long n)
{
  long i = (long)blockIdx.x * blockDim.x + threadIdx.x;
  if (i >= n) return;
  out[i] = f2bf(in[i]);
}

// ---------------------------------------------------------------------------
// A-matrix pack kernels (f32 activations -> bf16 GEMM rows)
// ---------------------------------------------------------------------------
// grouped k=2 conv: out (G, B*T, 256); row = [x[t-1, g-slice], x[t, g-slice]]
__global__ void pack_dis_kernel(bf16_t* __restrict__ out,
                                const float* __restrict__ in, int T)
{
  long idx = (long)blockIdx.x * blockDim.x + threadIdx.x;
  long M = (long)Bn * T;
  if (idx >= (long)Gg * M * 256) return;
  int c = (int)(idx & 255);
  long rem = idx >> 8;
  int m = (int)(rem % M);
  int g = (int)(rem / M);
  int s = c >> 7, i = c & 127;
  int b = m / T, t = m % T;
  int tt = t - 1 + s;
  float v = (tt < 0) ? 0.f : in[((long)b * T + tt) * Dm + g * 128 + i];
  out[idx] = f2bf(v);
}

// stride-2 full conv: out (B*T/2, 2048); row j = [x[2j-1], x[2j]]
__global__ void pack_iso_kernel(bf16_t* __restrict__ out,
                                const float* __restrict__ in, int T)
{
  long idx = (long)blockIdx.x * blockDim.x + threadIdx.x;
  int To = T >> 1;
  long total = (long)Bn * To * 2048;
  if (idx >= total) return;
  int c = (int)(idx & 2047);
  long m = idx >> 11;
  int b = (int)(m / To), j = (int)(m % To);
  int s = c >> 10, i = c & 1023;
  int tt = 2 * j - 1 + s;
  float v = (tt < 0) ? 0.f : in[((long)b * T + tt) * Dm + i];
  out[idx] = f2bf(v);
}

// repeat each timestep twice: out (B*2T, 1024)
__global__ void pack_rep_kernel(bf16_t* __restrict__ out,
                                const float* __restrict__ in, int Tin)
{
  long idx = (long)blockIdx.x * blockDim.x + threadIdx.x;
  int To = Tin * 2;
  long total = (long)Bn * To * Dm;
  if (idx >= total) return;
  int i = (int)(idx & 1023);
  long m = idx >> 10;
  int b = (int)(m / To), t = (int)(m % To);
  out[idx] = f2bf(in[((long)b * Tin + (t >> 1)) * Dm + i]);
}

// concat [a, sk]: out (M, 2048)
__global__ void pack_cat_kernel(bf16_t* __restrict__ out,
                                const float* __restrict__ a,
                                const float* __restrict__ sk, long M)
{
  long idx = (long)blockIdx.x * blockDim.x + threadIdx.x;
  if (idx >= M * 2048) return;
  int c = (int)(idx & 2047);
  long m = idx >> 11;
  float v = (c < Dm) ? a[m * Dm + c] : sk[m * Dm + (c - Dm)];
  out[idx] = f2bf(v);
}

// ---------------------------------------------------------------------------
// Weight pack kernels -> bf16 output-major (N, K)
// ---------------------------------------------------------------------------
// conv (O, I, 2) -> (O, 2I), col = s*I + i
__global__ void pack_wconv_kernel(bf16_t* __restrict__ out,
                                  const float* __restrict__ in, int O, int I)
{
  long idx = (long)blockIdx.x * blockDim.x + threadIdx.x;
  if (idx >= (long)O * 2 * I) return;
  int col = (int)(idx % (2 * I));
  long o = idx / (2 * I);
  int s = col / I, i = col % I;
  out[idx] = f2bf(in[(o * I + i) * 2 + s]);
}

// matmul W (K, N) -> Wt (N, K) transpose
__global__ void pack_wt_kernel(bf16_t* __restrict__ out,
                               const float* __restrict__ in, int Kd, int N)
{
  long idx = (long)blockIdx.x * blockDim.x + threadIdx.x;
  if (idx >= (long)Kd * N) return;
  int k = (int)(idx % Kd);
  long n = idx / Kd;
  out[idx] = f2bf(in[(long)k * N + n]);
}

// ---------------------------------------------------------------------------
// MPS scan: 32 blocks (b,h). h_t = tanh(u_t + h_{t-1} A_h-row), y = h C_h.
// A_h (256x256) staged in LDS with stride 257 (bank-conflict free).
// U layout (H, B*T, K); Y layout (B, T, D) with head-major columns.
// ---------------------------------------------------------------------------
__global__ __launch_bounds__(256)
void mps_scan_kernel(const float* __restrict__ U, const float* __restrict__ Amat,
                     const float* __restrict__ Cmat, float* __restrict__ Y, int T)
{
  extern __shared__ float lds[];
  float* sA = lds;                    // 256 * 257
  float* sH = lds + Kk * (Kk + 1);    // 256
  const int blk = blockIdx.x;
  const int b = blk / Hh, h = blk % Hh;
  const int t0 = threadIdx.x;

  const float* Ah = Amat + (long)h * Kk * Kk;
  for (int i = t0; i < Kk * Kk; i += 256) {
    int j = i >> 8, k = i & 255;
    sA[j * (Kk + 1) + k] = Ah[i];
  }
  sH[t0] = 0.f;
  __syncthreads();

  const float* Ch = Cmat + (long)h * DHh * Kk;
  const float* Ub = U + ((long)h * Bn * T + (long)b * T) * Kk;
  float* Yb = Y + (long)b * T * Dm + h * DHh;
  const float* arow = sA + t0 * (Kk + 1);

  for (int t = 0; t < T; ++t) {
    float z = Ub[(long)t * Kk + t0];
#pragma unroll 8
    for (int k = 0; k < Kk; ++k) z += sH[k] * arow[k];
    float hn = tanhf(z);
    __syncthreads();
    sH[t0] = hn;
    __syncthreads();
    if (t0 < DHh) {
      const float* crow = Ch + (long)t0 * Kk;
      float y = 0.f;
#pragma unroll 8
      for (int k = 0; k < Kk; ++k) y += sH[k] * crow[k];
      Yb[(long)t * Dm + t0] = y;
    }
  }
}

// ---------------------------------------------------------------------------
// Host orchestration
// ---------------------------------------------------------------------------
static inline long cdiv(long a, long b) { return (a + b - 1) / b; }

static void launch_gemm(hipStream_t s, const bf16_t* A, const bf16_t* Wt,
                        const float* bias, float* Cf, bf16_t* Cbf,
                        int M, int N, int Kd, int lda, int ldc,
                        long aZ, long bZ, long cZ, long biasZ, int Z, int epi)
{
  dim3 grid((unsigned)(N / BN), (unsigned)(M / BM), (unsigned)Z), blk(256);
  size_t shb = (size_t)2 * (BM + BN) * PADK * sizeof(bf16_t);  // 72 KB
  if (epi == 0)
    gemm_wmma<0><<<grid, blk, shb, s>>>(A, Wt, bias, Cf, Cbf, M, N, Kd, lda, ldc,
                                        aZ, bZ, cZ, biasZ);
  else
    gemm_wmma<1><<<grid, blk, shb, s>>>(A, Wt, bias, Cf, Cbf, M, N, Kd, lda, ldc,
                                        aZ, bZ, cZ, biasZ);
}

extern "C" void kernel_launch(void* const* d_in, const int* in_sizes, int n_in,
                              void* d_out, int out_size, void* d_ws, size_t ws_size,
                              hipStream_t stream)
{
  (void)in_sizes; (void)n_in; (void)out_size; (void)ws_size;
  // ---- input cursor (setup_inputs dict order, recursive insertion order) ----
  int ci = 0;
  auto nxt = [&]() { return (const float*)d_in[ci++]; };
  const float* x_in = nxt();
  struct DisP { const float *wg, *bg, *wv, *bv, *g, *b; };
  struct IsoP { const float *w, *b, *g, *bb; };
  struct InvP { const float *W, *b, *g, *bb; };
  struct SkpP { const float *W, *b; };
  DisP enc_dis[3], dec_dis[3];
  IsoP enc_iso[3];
  InvP dec_inv[3];
  SkpP skp[3];
  for (int i = 0; i < 3; ++i) enc_dis[i] = {nxt(), nxt(), nxt(), nxt(), nxt(), nxt()};
  for (int i = 0; i < 3; ++i) enc_iso[i] = {nxt(), nxt(), nxt(), nxt()};
  for (int i = 0; i < 3; ++i) dec_inv[i] = {nxt(), nxt(), nxt(), nxt()};
  for (int i = 0; i < 3; ++i) dec_dis[i] = {nxt(), nxt(), nxt(), nxt(), nxt(), nxt()};
  for (int i = 0; i < 3; ++i) skp[i] = {nxt(), nxt()};
  const float *mpsA = nxt(), *mpsB = nxt(), *mpsC = nxt(), *mpsWo = nxt(),
              *mpsbo = nxt(), *mpsg = nxt(), *mpsb = nxt();
  const float *W1 = nxt(), *b1 = nxt(), *W2 = nxt(), *b2 = nxt();
  const float *n1g = nxt(), *n1b = nxt(), *n2g = nxt(), *n2b = nxt();

  // ---- workspace bump allocator ----
  char* ws = (char*)d_ws;
  size_t off = 0;
  auto alloc = [&](size_t bytes) {
    void* p = ws + off;
    off = (off + bytes + 255) & ~(size_t)255;
    return p;
  };
  const long M0 = (long)Bn * T0;                    // 16384
  float* P = (float*)alloc(M0 * Dm * 4);
  float* Q = (float*)alloc(M0 * Dm * 4);
  float* R = (float*)alloc(M0 * Dm * 4);
  float* skbuf[3];
  skbuf[0] = (float*)alloc(M0 * Dm * 4);
  skbuf[1] = (float*)alloc(M0 / 2 * Dm * 4);
  skbuf[2] = (float*)alloc(M0 / 4 * Dm * 4);
  float* Ubuf = (float*)alloc((long)Hh * Bn * 512 * Kk * 4);
  bf16_t* PACKA = (bf16_t*)alloc((size_t)M0 * 2048 * 2);        // 64 MB
  bf16_t* PACKB = (bf16_t*)alloc((size_t)M0 * DFFc * 2);        // 128 MB
  bf16_t *wgTe[3], *wvTe[3], *wgTd[3], *wvTd[3], *isoT[3], *invT[3], *skpT[3];
  for (int i = 0; i < 3; ++i) {
    wgTe[i] = (bf16_t*)alloc((size_t)Dm * 256 * 2);
    wvTe[i] = (bf16_t*)alloc((size_t)Dm * 256 * 2);
    wgTd[i] = (bf16_t*)alloc((size_t)Dm * 256 * 2);
    wvTd[i] = (bf16_t*)alloc((size_t)Dm * 256 * 2);
    isoT[i] = (bf16_t*)alloc((size_t)Dm * 2048 * 2);
    invT[i] = (bf16_t*)alloc((size_t)Dm * Dm * 2);
    skpT[i] = (bf16_t*)alloc((size_t)Dm * 2048 * 2);
  }
  bf16_t* BwT = (bf16_t*)alloc((size_t)Hh * Kk * DHh * 2);
  bf16_t* WoT = (bf16_t*)alloc((size_t)Dm * Dm * 2);
  bf16_t* W1T = (bf16_t*)alloc((size_t)DFFc * Dm * 2);
  bf16_t* W2T = (bf16_t*)alloc((size_t)Dm * DFFc * 2);

  auto g1 = [&](long n) { return dim3((unsigned)cdiv(n, 256)); };

  // ---- weight packing (bf16, output-major) ----
  for (int i = 0; i < 3; ++i) {
    long nc = (long)Dm * 256;
    pack_wconv_kernel<<<g1(nc), 256, 0, stream>>>(wgTe[i], enc_dis[i].wg, Dm, 128);
    pack_wconv_kernel<<<g1(nc), 256, 0, stream>>>(wvTe[i], enc_dis[i].wv, Dm, 128);
    pack_wconv_kernel<<<g1(nc), 256, 0, stream>>>(wgTd[i], dec_dis[i].wg, Dm, 128);
    pack_wconv_kernel<<<g1(nc), 256, 0, stream>>>(wvTd[i], dec_dis[i].wv, Dm, 128);
    pack_wconv_kernel<<<g1((long)Dm * 2048), 256, 0, stream>>>(isoT[i], enc_iso[i].w, Dm, Dm);
    pack_wt_kernel<<<g1((long)Dm * Dm), 256, 0, stream>>>(invT[i], dec_inv[i].W, Dm, Dm);
    pack_wt_kernel<<<g1((long)2048 * Dm), 256, 0, stream>>>(skpT[i], skp[i].W, 2048, Dm);
  }
  cvt_bf_kernel<<<g1((long)Hh * Kk * DHh), 256, 0, stream>>>(BwT, mpsB, (long)Hh * Kk * DHh);
  pack_wt_kernel<<<g1((long)Dm * Dm), 256, 0, stream>>>(WoT, mpsWo, Dm, Dm);
  pack_wt_kernel<<<g1((long)Dm * DFFc), 256, 0, stream>>>(W1T, W1, Dm, DFFc);
  pack_wt_kernel<<<g1((long)DFFc * Dm), 256, 0, stream>>>(W2T, W2, DFFc, Dm);

  // ---- forward ----
  // x = LN(x_in) -> P
  ln_kernel<<<(unsigned)M0, 256, 0, stream>>>(P, x_in, nullptr, n1g, n1b);
  float* cur = P;
  int T = T0;

  // encoder
  for (int i = 0; i < 3; ++i) {
    long M = (long)Bn * T;
    pack_dis_kernel<<<g1((long)Gg * M * 256), 256, 0, stream>>>(PACKA, cur, T);
    launch_gemm(stream, PACKA, wgTe[i], enc_dis[i].bg, Q, nullptr,
                (int)M, 128, 256, 256, Dm, M * 256, 128 * 256, 128, 128, Gg, 0);
    launch_gemm(stream, PACKA, wvTe[i], enc_dis[i].bv, R, nullptr,
                (int)M, 128, 256, 256, Dm, M * 256, 128 * 256, 128, 128, Gg, 0);
    sigmoid_mul_kernel<<<g1(M * Dm), 256, 0, stream>>>(Q, Q, R, M * Dm);
    ln_kernel<<<(unsigned)M, 256, 0, stream>>>(skbuf[i], Q, nullptr,
                                               enc_dis[i].g, enc_dis[i].b);
    // isometry (stride-2 conv)
    pack_iso_kernel<<<g1(M / 2 * 2048), 256, 0, stream>>>(PACKA, skbuf[i], T);
    launch_gemm(stream, PACKA, isoT[i], enc_iso[i].b, Q, nullptr,
                (int)(M / 2), Dm, 2048, 2048, Dm, 0, 0, 0, 0, 1, 0);
    ln_kernel<<<(unsigned)(M / 2), 256, 0, stream>>>(P, Q, nullptr,
                                                     enc_iso[i].g, enc_iso[i].bb);
    cur = P;
    T >>= 1;
  }

  // MPS (T = 512)
  const long M3 = (long)Bn * T;
  cvt_bf_kernel<<<g1(M3 * Dm), 256, 0, stream>>>(PACKA, cur, M3 * Dm);
  // u[h] = X_h @ B_h^T : A z-offset selects head columns (aZ = 128)
  launch_gemm(stream, PACKA, BwT, nullptr, Ubuf, nullptr,
              (int)M3, Kk, DHh, Dm, Kk, 128, (long)Kk * DHh, M3 * Kk, 0, Hh, 0);
  {
    size_t ldsB = ((size_t)Kk * (Kk + 1) + Kk) * 4;  // ~258 KB < 320 KB/WGP
    mps_scan_kernel<<<(unsigned)(Bn * Hh), 256, ldsB, stream>>>(Ubuf, mpsA, mpsC, Q, T);
  }
  cvt_bf_kernel<<<g1(M3 * Dm), 256, 0, stream>>>(PACKA, Q, M3 * Dm);
  launch_gemm(stream, PACKA, WoT, mpsbo, R, nullptr,
              (int)M3, Dm, Dm, Dm, Dm, 0, 0, 0, 0, 1, 0);
  ln_kernel<<<(unsigned)M3, 256, 0, stream>>>(Q, R, cur, mpsg, mpsb);
  cur = Q;

  // decoder
  for (int i = 2; i >= 0; --i) {
    int Tout = T * 2;
    long Mout = (long)Bn * Tout;
    float* o1 = (cur == P) ? Q : P;
    pack_rep_kernel<<<g1(Mout * Dm), 256, 0, stream>>>(PACKA, cur, T);
    launch_gemm(stream, PACKA, invT[i], dec_inv[i].b, o1, nullptr,
                (int)Mout, Dm, Dm, Dm, Dm, 0, 0, 0, 0, 1, 0);
    ln_kernel<<<(unsigned)Mout, 256, 0, stream>>>(o1, o1, nullptr,
                                                  dec_inv[i].g, dec_inv[i].bb);
    pack_cat_kernel<<<g1(Mout * 2048), 256, 0, stream>>>(PACKA, o1, skbuf[i], Mout);
    launch_gemm(stream, PACKA, skpT[i], skp[i].b, R, nullptr,
                (int)Mout, Dm, 2048, 2048, Dm, 0, 0, 0, 0, 1, 0);
    pack_dis_kernel<<<g1((long)Gg * Mout * 256), 256, 0, stream>>>(PACKA, R, Tout);
    launch_gemm(stream, PACKA, wgTd[i], dec_dis[i].bg, o1, nullptr,
                (int)Mout, 128, 256, 256, Dm, Mout * 256, 128 * 256, 128, 128, Gg, 0);
    launch_gemm(stream, PACKA, wvTd[i], dec_dis[i].bv, cur, nullptr,
                (int)Mout, 128, 256, 256, Dm, Mout * 256, 128 * 256, 128, 128, Gg, 0);
    sigmoid_mul_kernel<<<g1(Mout * Dm), 256, 0, stream>>>(o1, o1, cur, Mout * Dm);
    ln_kernel<<<(unsigned)Mout, 256, 0, stream>>>(o1, o1, nullptr,
                                                  dec_dis[i].g, dec_dis[i].b);
    cur = o1;
    T = Tout;
  }

  // final residual + FFN
  float* xres = (cur == P) ? Q : P;
  add2_kernel<<<g1(M0 * Dm), 256, 0, stream>>>(xres, x_in, cur, M0 * Dm);
  ln_kernel<<<(unsigned)M0, 256, 0, stream>>>(R, xres, nullptr, n2g, n2b);
  cvt_bf_kernel<<<g1(M0 * Dm), 256, 0, stream>>>(PACKA, R, M0 * Dm);
  // h = gelu(x @ W1 + b1): bf16 output straight into GEMM2's A
  launch_gemm(stream, PACKA, W1T, b1, nullptr, PACKB,
              (int)M0, DFFc, Dm, Dm, DFFc, 0, 0, 0, 0, 1, 1);
  launch_gemm(stream, PACKB, W2T, b2, R, nullptr,
              (int)M0, Dm, DFFc, DFFc, Dm, 0, 0, 0, 0, 1, 0);
  add2_kernel<<<g1(M0 * Dm), 256, 0, stream>>>((float*)d_out, xres, R, M0 * Dm);
}
